// My_Conv_block_6597069766883
// MI455X (gfx1250) — matrix operand order repaired and checked by reading, not compiled, not run on hardware
//
#include <hip/hip_runtime.h>
#include <hip/hip_bf16.h>

typedef __attribute__((ext_vector_type(16))) _Float16 v16h;
typedef __attribute__((ext_vector_type(8)))  _Float16 v8h;
typedef __attribute__((ext_vector_type(8)))  float    v8f;

#define NB   4
#define C2   128
#define HW   96
#define NPIX (HW*HW)          // 9216
#define TPI  (NPIX/16)        // 576 n-tiles per image
#define NT   (NB*TPI)         // 2304 n-tiles total
#define KDEF (C2*9)           // 1152
#define BROW 24               // padded LDS row stride (halfs) per (group,n)

// ---- WMMA fragment loaders -------------------------------------------------
// A (16x32 f16): lane holds row M=lane&15; halves 0..15 at K=kc+2*sel..+15,
// sel = (lane>=16)*8.
__device__ __forceinline__ v16h ld_a(const _Float16* wrow, int kc, int sel) {
    union { v16h v; v8h h[2]; } u;
    u.h[0] = *(const v8h*)(wrow + kc + 2*sel);
    u.h[1] = *(const v8h*)(wrow + kc + 2*sel + 8);
    return u.v;
}
// B (32x16 f16) from staged LDS: group g = 2c + (lane>=16), col n = lane&15,
// 16 contiguous K values at row start.
__device__ __forceinline__ v16h ld_b(const _Float16* lds, int c, int lane) {
    const _Float16* p = lds + (size_t)((2*c + (lane>>4))*16 + (lane & 15)) * BROW;
    union { v16h v; v8h h[2]; } u;
    u.h[0] = *(const v8h*)p;
    u.h[1] = *(const v8h*)(p + 8);
    return u.v;
}
__device__ __forceinline__ v8f wmma_f16(v16h a, v16h b, v8f c) {
    return __builtin_amdgcn_wmma_f32_16x16x32_f16(false, a, false, b, (short)0, c, false, false);
}
// staged LDS write index for element (K, n)
__device__ __forceinline__ int bidx(int K, int n) {
    return ((K >> 4)*16 + n) * BROW + (K & 15);
}

// ---- weight f32 -> f16 -----------------------------------------------------
__global__ void k_cvt_f16(const float* __restrict__ in, _Float16* __restrict__ out, int n) {
    int i = blockIdx.x*256 + threadIdx.x;
    if (i < n) out[i] = (_Float16)in[i];
}

// ---- kernel A: 1x1 conv x -> x_c (M=128,K=128) -----------------------------
__global__ void k_conv1x1_in(const float* __restrict__ x, const _Float16* __restrict__ wh,
                             const float* __restrict__ bias, float* __restrict__ xc) {
    __shared__ _Float16 lb[8*16*BROW];
    int tile = blockIdx.x, b = tile / TPI, p = (tile % TPI)*16;
    int tid = threadIdx.x;
    const float* xb = x + (size_t)b*C2*NPIX + p;
    #pragma unroll
    for (int i = 0; i < 8; ++i) {
        int idx = tid + i*256, K = idx >> 4, n = idx & 15;
        lb[bidx(K, n)] = (_Float16)xb[(size_t)K*NPIX + n];
    }
    __syncthreads();
    int lane = tid & 31, wave = tid >> 5;
    int cobase = wave*16, sel = (lane >> 4)*8;
    const _Float16* wrow = wh + (size_t)(cobase + (lane & 15))*C2;
    v8f acc = {};
    #pragma unroll
    for (int c = 0; c < 4; ++c)
        acc = wmma_f16(ld_a(wrow, 32*c, sel), ld_b(lb, c, lane), acc);
    float* op = xc + (size_t)b*C2*NPIX + p + (lane & 15);
    #pragma unroll
    for (int r = 0; r < 8; ++r) {
        int co = cobase + r + ((lane >> 4) ? 8 : 0);
        op[(size_t)co*NPIX] = acc[r] + bias[co];
    }
}

// ---- kernel B: 3x3 offset conv (18 ch) -------------------------------------
__global__ void k_offset(const float* __restrict__ xc, const float* __restrict__ wo,
                         const float* __restrict__ bo, float* __restrict__ off) {
    int idx = blockIdx.x*256 + threadIdx.x;
    if (idx >= NB*18*NPIX) return;
    int w = idx % HW, h = (idx/HW) % HW, oc = (idx/NPIX) % 18, b = idx/(18*NPIX);
    float acc = bo[oc];
    const float* xb = xc + (size_t)b*C2*NPIX;
    const float* wb = wo + (size_t)oc*C2*9;
    for (int ci = 0; ci < C2; ++ci) {
        const float* xi = xb + (size_t)ci*NPIX;
        const float* wi = wb + ci*9;
        #pragma unroll
        for (int ki = 0; ki < 3; ++ki) {
            int hh = h + ki - 1;
            if (hh < 0 || hh >= HW) continue;
            #pragma unroll
            for (int kj = 0; kj < 3; ++kj) {
                int ww = w + kj - 1;
                if (ww < 0 || ww >= HW) continue;
                acc = fmaf(wi[ki*3+kj], xi[hh*HW + ww], acc);
            }
        }
    }
    off[idx] = acc;
}

// ---- kernel C: deformable branch (gather + GEMM K=1152 + BN/ReLU) ----------
__global__ void k_deform(const float* __restrict__ xc, const float* __restrict__ off,
                         const _Float16* __restrict__ wdh, const float* __restrict__ bd,
                         const float* __restrict__ bn1, _Float16* __restrict__ cat) {
    __shared__ _Float16 lb[72*16*BROW];
    __shared__ float cw[144*4];
    __shared__ int   cx[144*4];
    int tile = blockIdx.x, b = tile / TPI, p = (tile % TPI)*16;
    int h = p / HW, w0 = p % HW;
    int tid = threadIdx.x;
    if (tid < 144) {
        int pix = tid / 9, k = tid % 9, w = w0 + pix;
        float ox = off[((size_t)(b*18 + k))*NPIX + h*HW + w];
        float oy = off[((size_t)(b*18 + 9 + k))*NPIX + h*HW + w];
        float px = ox + (float)(k/3 - 1) + (float)(h + 1);
        float py = oy + (float)(k%3 - 1) + (float)(w + 1);
        float qx = floorf(px), qy = floorf(py);
        float ltx = fminf(fmaxf(qx,       0.f), 97.f);
        float lty = fminf(fmaxf(qy,       0.f), 97.f);
        float rbx = fminf(fmaxf(qx + 1.f, 0.f), 97.f);
        float rby = fminf(fmaxf(qy + 1.f, 0.f), 97.f);
        float pxc = fminf(fmaxf(px, 0.f), 97.f);
        float pyc = fminf(fmaxf(py, 0.f), 97.f);
        float gx0 = 1.f + ltx - pxc, gx1 = 1.f - rbx + pxc;
        float gy0 = 1.f + lty - pyc, gy1 = 1.f - rby + pyc;
        float gs[4] = {gx0*gy0, gx1*gy1, gx0*gy1, gx1*gy0};
        float xs[4] = {ltx, rbx, ltx, rbx};
        float ys[4] = {lty, rby, rby, lty};
        #pragma unroll
        for (int c = 0; c < 4; ++c) {
            int ix = (int)xs[c], iy = (int)ys[c];
            bool v = (ix >= 1) & (ix <= 96) & (iy >= 1) & (iy <= 96);
            cw[tid*4 + c] = v ? gs[c] : 0.f;
            cx[tid*4 + c] = v ? ((ix - 1)*HW + (iy - 1)) : 0;
        }
    }
    __syncthreads();
    for (int e = tid; e < KDEF*16; e += 256) {
        int pix = e & 15, K = e >> 4;
        int ci = K / 9, k = K - ci*9, j = (pix*9 + k)*4;
        const float* base = xc + ((size_t)(b*C2 + ci))*NPIX;
        float v = cw[j+0]*base[cx[j+0]] + cw[j+1]*base[cx[j+1]]
                + cw[j+2]*base[cx[j+2]] + cw[j+3]*base[cx[j+3]];
        lb[bidx(K, pix)] = (_Float16)v;
    }
    __syncthreads();
    int lane = tid & 31, wave = tid >> 5;
    int cobase = wave*16, sel = (lane >> 4)*8;
    const _Float16* wrow = wdh + (size_t)(cobase + (lane & 15))*KDEF;
    v8f acc = {};
    for (int c = 0; c < 36; ++c)
        acc = wmma_f16(ld_a(wrow, 32*c, sel), ld_b(lb, c, lane), acc);
    _Float16* op = cat + (size_t)b*(2*C2)*NPIX + p + (lane & 15);
    #pragma unroll
    for (int r = 0; r < 8; ++r) {
        int co = cobase + r + ((lane >> 4) ? 8 : 0);
        float inv = bn1[co] / sqrtf(bn1[3*C2 + co] + 1e-5f);
        float y = (acc[r] + bd[co]) * inv + (bn1[C2 + co] - bn1[2*C2 + co]*inv);
        op[(size_t)co*NPIX] = (_Float16)fmaxf(y, 0.f);
    }
}

// ---- kernel D: standard 3x3 branch (im2col GEMM + BN/ReLU + residual) ------
__global__ void k_std(const float* __restrict__ xc, const _Float16* __restrict__ wsh,
                      const float* __restrict__ bs, const float* __restrict__ bn2,
                      _Float16* __restrict__ cat) {
    __shared__ _Float16 lb[72*16*BROW];
    int tile = blockIdx.x, b = tile / TPI, p = (tile % TPI)*16;
    int h = p / HW, w0 = p % HW;
    int tid = threadIdx.x;
    for (int e = tid; e < KDEF*16; e += 256) {
        int pix = e & 15, K = e >> 4;
        int ci = K / 9, k = K - ci*9;
        int hh = h + k/3 - 1, ww = w0 + pix + (k%3) - 1;
        float v = 0.f;
        if (hh >= 0 && hh < HW && ww >= 0 && ww < HW)
            v = xc[((size_t)(b*C2 + ci))*NPIX + hh*HW + ww];
        lb[bidx(K, pix)] = (_Float16)v;
    }
    __syncthreads();
    int lane = tid & 31, wave = tid >> 5;
    int cobase = wave*16, sel = (lane >> 4)*8;
    const _Float16* wrow = wsh + (size_t)(cobase + (lane & 15))*KDEF;
    v8f acc = {};
    for (int c = 0; c < 36; ++c)
        acc = wmma_f16(ld_a(wrow, 32*c, sel), ld_b(lb, c, lane), acc);
    int n = lane & 15;
    _Float16* op = cat + (size_t)b*(2*C2)*NPIX + p + n;
    const float* res = xc + (size_t)b*C2*NPIX + p + n;
    #pragma unroll
    for (int r = 0; r < 8; ++r) {
        int co = cobase + r + ((lane >> 4) ? 8 : 0);
        float inv = bn2[co] / sqrtf(bn2[3*C2 + co] + 1e-5f);
        float y = (acc[r] + bs[co]) * inv + (bn2[C2 + co] - bn2[2*C2 + co]*inv);
        y = fmaxf(y, 0.f) + res[(size_t)co*NPIX];
        op[(size_t)(C2 + co)*NPIX] = (_Float16)y;
    }
}

// ---- kernel E: final 1x1 fuse (M=256,K=256), 512 threads = 16 waves --------
__global__ void k_conv1x1_out(const _Float16* __restrict__ cat, const _Float16* __restrict__ wh,
                              const float* __restrict__ bias, float* __restrict__ out) {
    __shared__ _Float16 lb[16*16*BROW];
    int tile = blockIdx.x, b = tile / TPI, p = (tile % TPI)*16;
    int tid = threadIdx.x;
    const _Float16* cb = cat + (size_t)b*256*NPIX + p;
    #pragma unroll
    for (int i = 0; i < 8; ++i) {
        int idx = tid + i*512, K = idx >> 4, n = idx & 15;
        lb[bidx(K, n)] = cb[(size_t)K*NPIX + n];
    }
    __syncthreads();
    int lane = tid & 31, wave = tid >> 5;   // 0..15
    int cobase = wave*16, sel = (lane >> 4)*8;
    const _Float16* wrow = wh + (size_t)(cobase + (lane & 15))*256;
    v8f acc = {};
    #pragma unroll
    for (int c = 0; c < 8; ++c)
        acc = wmma_f16(ld_a(wrow, 32*c, sel), ld_b(lb, c, lane), acc);
    float* op = out + (size_t)b*256*NPIX + p + (lane & 15);
    #pragma unroll
    for (int r = 0; r < 8; ++r) {
        int co = cobase + r + ((lane >> 4) ? 8 : 0);
        op[(size_t)co*NPIX] = acc[r] + bias[co];
    }
}

// ---- workspace layout (bytes) ----------------------------------------------
#define O_XC   ((size_t)0)                          // f32  4*128*9216
#define O_OFF  (O_XC  + (size_t)NB*C2*NPIX*4)       // f32  4*18*9216
#define O_CAT  (O_OFF + (size_t)NB*18*NPIX*4)       // f16  4*256*9216
#define O_WIN  (O_CAT + (size_t)NB*256*NPIX*2)      // f16  128*128
#define O_WDEF (O_WIN + (size_t)C2*C2*2)            // f16  128*1152
#define O_WSTD (O_WDEF + (size_t)C2*KDEF*2)         // f16  128*1152
#define O_WOUT (O_WSTD + (size_t)C2*KDEF*2)         // f16  256*256

extern "C" void kernel_launch(void* const* d_in, const int* in_sizes, int n_in,
                              void* d_out, int out_size, void* d_ws, size_t ws_size,
                              hipStream_t stream) {
    const float* x     = (const float*)d_in[0];
    const float* w_in  = (const float*)d_in[1];
    const float* b_in  = (const float*)d_in[2];
    const float* w_off = (const float*)d_in[3];
    const float* b_off = (const float*)d_in[4];
    const float* w_def = (const float*)d_in[5];
    const float* b_def = (const float*)d_in[6];
    const float* bn1   = (const float*)d_in[7];
    const float* w_std = (const float*)d_in[8];
    const float* b_std = (const float*)d_in[9];
    const float* bn2   = (const float*)d_in[10];
    const float* w_out = (const float*)d_in[11];
    const float* b_out = (const float*)d_in[12];

    char* ws = (char*)d_ws;
    float*     xc   = (float*)(ws + O_XC);
    float*     offb = (float*)(ws + O_OFF);
    _Float16*  cat  = (_Float16*)(ws + O_CAT);
    _Float16*  winh = (_Float16*)(ws + O_WIN);
    _Float16*  wdh  = (_Float16*)(ws + O_WDEF);
    _Float16*  wsh  = (_Float16*)(ws + O_WSTD);
    _Float16*  woh  = (_Float16*)(ws + O_WOUT);

    k_cvt_f16<<<(C2*C2 + 255)/256, 256, 0, stream>>>(w_in, winh, C2*C2);
    k_cvt_f16<<<(C2*KDEF + 255)/256, 256, 0, stream>>>(w_def, wdh, C2*KDEF);
    k_cvt_f16<<<(C2*KDEF + 255)/256, 256, 0, stream>>>(w_std, wsh, C2*KDEF);
    k_cvt_f16<<<(256*256 + 255)/256, 256, 0, stream>>>(w_out, woh, 256*256);

    k_conv1x1_in<<<NT, 256, 0, stream>>>(x, winh, b_in, xc);
    k_offset<<<(NB*18*NPIX + 255)/256, 256, 0, stream>>>(xc, w_off, b_off, offb);
    k_deform<<<NT, 256, 0, stream>>>(xc, offb, wdh, b_def, bn1, cat);
    k_std<<<NT, 256, 0, stream>>>(xc, wsh, b_std, bn2, cat);
    k_conv1x1_out<<<NT, 512, 0, stream>>>(cat, woh, b_out, (float*)d_out);
}
